// Gate_net_20478404067558
// MI455X (gfx1250) — matrix operand order repaired and checked by reading, not compiled, not run on hardware
//
#include <hip/hip_runtime.h>

#define N_NODES      120000
#define N_PER_LEVEL  7500
#define HALF_LVL     3750
#define N_LEVELS     16
#define D            128
#define EDGES_PER_LVL 11250

#define XSTR 264   // AND gather tile row stride (ushorts), K up to 256 + pad
#define VSTR 132   // vout row stride (floats)
#define MSTR 136   // msg tile row stride (ushorts)
#define NSTR 136   // NOT tiles row stride (ushorts)

typedef __attribute__((ext_vector_type(16))) __bf16 v16bf;
typedef __attribute__((ext_vector_type(8)))  float  v8f;

__device__ __forceinline__ unsigned short f2bf(float f) {
  union { float f; unsigned u; } c; c.f = f;
  unsigned u = c.u;
  return (unsigned short)((u + 0x7FFFu + ((u >> 16) & 1u)) >> 16);
}

__device__ __forceinline__ float sigmoidf_(float x) { return 1.0f / (1.0f + expf(-x)); }

// A-fragment (16x32 bf16) from row-major bf16 LDS tile.
// lanes 0-15: v0..3 -> K=2v,2v+1 ; v4..7 -> K=16+2(v-4)+{0,1}; lanes 16-31: +8
__device__ __forceinline__ v16bf load_a_frag(const unsigned short* __restrict__ tile,
                                             int stride, int rowbase, int kbase, int lane) {
  const int m  = rowbase + (lane & 15);
  const int g8 = (lane >> 4) << 3;
  union { v16bf v; unsigned u[8]; } r;
  const unsigned short* row = tile + (size_t)m * stride + kbase;
#pragma unroll
  for (int v = 0; v < 8; ++v) {
    const int k = (v < 4) ? (v * 2 + g8) : (16 + (v - 4) * 2 + g8);
    r.u[v] = *(const unsigned*)(row + k);
  }
  return r.v;
}

// B-fragment: packed contiguously per (kt,nt,lane) by pack_b_kernel
__device__ __forceinline__ v16bf load_b_frag(const unsigned short* __restrict__ Bp,
                                             int ntiles, int kt, int nt, int lane) {
  const unsigned short* p = Bp + (((size_t)kt * ntiles + nt) * 32 + lane) * 16;
  return *(const v16bf*)p;
}

__device__ __forceinline__ v8f wmma_bf16(v16bf a, v16bf b, v8f c) {
#if defined(__gfx1250__)
  return __builtin_amdgcn_wmma_f32_16x16x32_bf16(false, a, false, b, (short)0, c, false, false);
#else
  return c;
#endif
}

// ---------------- precompute kernels ----------------

// Pack W[out,in] (f32 row-major) into WMMA-B bf16 fragments.
// frag = kt*ntiles + nt; element e -> k = kt*32 + (lane>>4)*8 + (e<8 ? e : 16+e-8), n = nt*16 + lane%16
__global__ void pack_b_kernel(const float* __restrict__ W, unsigned short* __restrict__ Bp,
                              int out_d, int in_d) {
  const int ntiles = out_d >> 4;
  const int frag = blockIdx.x;
  const int lane = threadIdx.x;
  const int kt = frag / ntiles, nt = frag % ntiles;
  const int n = nt * 16 + (lane & 15);
  const int kbase = kt * 32 + (lane >> 4) * 8;
  unsigned short* dst = Bp + ((size_t)frag * 32 + lane) * 16;
#pragma unroll
  for (int e = 0; e < 16; ++e)
    dst[e] = f2bf(W[(size_t)n * in_d + kbase + (e < 8 ? e : e + 8)]);
}

__global__ void hf0_kernel(const float* __restrict__ w, const float* __restrict__ b,
                           float* __restrict__ hf0) {
  const int c = threadIdx.x;
  hf0[c] = w[c] + b[c];   // hf_init: W[128,1] @ [1] + b
}

__global__ void init_state_kernel(float* __restrict__ hs, float* __restrict__ hf,
                                  const float* __restrict__ hf0) {
  const size_t i = (size_t)blockIdx.x * 256 + threadIdx.x;
  const int c = (int)(i & (D - 1));
  hs[i] = 0.0f;
  hf[i] = hf0[c];
}

// u[c] = sum_j Wk[j, c] * wa[128 + j]  (folds attention k-projection into one vector)
__global__ void u_kernel(const float* __restrict__ Wk, const float* __restrict__ wa,
                         float* __restrict__ u, int in_d) {
  const int c = blockIdx.x * 64 + threadIdx.x;
  if (c >= in_d) return;
  float s = 0.0f;
  for (int j = 0; j < D; ++j) s += Wk[(size_t)j * in_d + c] * wa[128 + j];
  u[c] = s;
}

// gh[o] = b_hh[o] (+ W_hh[o,:]·h0 if h0 != null)
__global__ void ghpre_kernel(const float* __restrict__ Whh, const float* __restrict__ bhh,
                             const float* __restrict__ h0, float* __restrict__ out) {
  const int o = blockIdx.x * 128 + threadIdx.x;
  if (o >= 384) return;
  float s = bhh[o];
  if (h0) {
    for (int c = 0; c < D; ++c) s += Whh[(size_t)o * D + c] * h0[c];
  }
  out[o] = s;
}

// ---------------- shared GEMM helpers ----------------

// V = X @ Wv^T + bv over BOTH 16-row tiles; each B fragment loaded once,
// shared by the two row-tile accumulators.
template<int KT>
__device__ __forceinline__ void vgemm_both(
    const unsigned short* __restrict__ xt,
    const unsigned short* __restrict__ Pv, const float* __restrict__ bv,
    float* __restrict__ vout, int lane) {
  v16bf a0[KT], a1[KT];
#pragma unroll
  for (int kt = 0; kt < KT; ++kt) a0[kt] = load_a_frag(xt, XSTR, 0,  kt * 32, lane);
#pragma unroll
  for (int kt = 0; kt < KT; ++kt) a1[kt] = load_a_frag(xt, XSTR, 16, kt * 32, lane);
  const int g8 = (lane >> 4) << 3;
  const int nlo = lane & 15;
#pragma unroll
  for (int nt = 0; nt < 8; ++nt) {
    v8f c0 = {0,0,0,0,0,0,0,0}, c1 = {0,0,0,0,0,0,0,0};
#pragma unroll
    for (int kt = 0; kt < KT; ++kt) {
      const v16bf b = load_b_frag(Pv, 8, kt, nt, lane);
      c0 = wmma_bf16(a0[kt], b, c0);
      c1 = wmma_bf16(a1[kt], b, c1);
    }
    const int col = nt * 16 + nlo;
    const float bb = bv[col];
#pragma unroll
    for (int v = 0; v < 8; ++v) {
      vout[(size_t)(v + g8) * VSTR + col]      = c0[v] + bb;
      vout[(size_t)(16 + v + g8) * VSTR + col] = c1[v] + bb;
    }
  }
}

// gx = msg @ W_ih^T; fused GRU epilogue; h = h0[col] (or 0); writes dst rows
__device__ __forceinline__ void gru_write(
    const unsigned short* __restrict__ msgt, int mstride,
    const unsigned short* __restrict__ Pih, const float* __restrict__ bih,
    const float* __restrict__ gh, const float* __restrict__ h0,
    float* __restrict__ dst, int node0, int nvalid, int lane) {
  v16bf af[4];
#pragma unroll
  for (int kt = 0; kt < 4; ++kt)
    af[kt] = load_a_frag(msgt, mstride, 0, kt * 32, lane);
  const int g8 = (lane >> 4) << 3;
  const int nlo = lane & 15;
#pragma unroll
  for (int nt = 0; nt < 8; ++nt) {
    v8f ar = {0,0,0,0,0,0,0,0}, az = {0,0,0,0,0,0,0,0}, an = {0,0,0,0,0,0,0,0};
#pragma unroll
    for (int kt = 0; kt < 4; ++kt) {
      ar = wmma_bf16(af[kt], load_b_frag(Pih, 24, kt, nt,      lane), ar);
      az = wmma_bf16(af[kt], load_b_frag(Pih, 24, kt, nt + 8,  lane), az);
      an = wmma_bf16(af[kt], load_b_frag(Pih, 24, kt, nt + 16, lane), an);
    }
    const int col = nt * 16 + nlo;
    const float br = bih[col]       + gh[col];
    const float bz = bih[col + 128] + gh[col + 128];
    const float bn = bih[col + 256];
    const float gn = gh[col + 256];
    const float h  = h0 ? h0[col] : 0.0f;
#pragma unroll
    for (int v = 0; v < 8; ++v) {
      const int m = v + g8;
      const float r  = sigmoidf_(ar[v] + br);
      const float z  = sigmoidf_(az[v] + bz);
      const float nn = tanhf(an[v] + bn + r * gn);
      const float hn = (1.0f - z) * nn + z * h;
      if (m < nvalid)
        dst[(size_t)(node0 + m) * D + col] = hn;
    }
  }
}

// ---------------- per-level kernels ----------------

template<bool ISF>
__device__ __forceinline__ void and_phase(
    int lane, int j0, int node0, int src,
    float* __restrict__ hs, float* __restrict__ hf, const float* __restrict__ u,
    const unsigned short* __restrict__ Pv, const float* __restrict__ bv,
    const unsigned short* __restrict__ Pih, const float* __restrict__ bih,
    const float* __restrict__ gh, const float* __restrict__ hf0,
    unsigned short* __restrict__ xt, float* __restrict__ vout,
    unsigned short* __restrict__ msgt, float* __restrict__ sc, float* __restrict__ wts) {
  constexpr int KT = ISF ? 8 : 4;
  // gather x[src] row (bf16) + attention score u·x
  {
    float s = 0.0f;
    unsigned short* xrow = xt + (size_t)lane * XSTR;
    const float* p0 = hs + (size_t)src * D;
#pragma unroll 4
    for (int c = 0; c < D; c += 4) {
      float4 v = *(const float4*)(p0 + c);
      xrow[c] = f2bf(v.x); xrow[c+1] = f2bf(v.y); xrow[c+2] = f2bf(v.z); xrow[c+3] = f2bf(v.w);
      s += v.x*u[c] + v.y*u[c+1] + v.z*u[c+2] + v.w*u[c+3];
    }
    if (ISF) {
      const float* p1 = hf + (size_t)src * D;
#pragma unroll 4
      for (int c = 0; c < D; c += 4) {
        float4 v = *(const float4*)(p1 + c);
        xrow[D+c] = f2bf(v.x); xrow[D+c+1] = f2bf(v.y); xrow[D+c+2] = f2bf(v.z); xrow[D+c+3] = f2bf(v.w);
        s += v.x*u[D+c] + v.y*u[D+c+1] + v.z*u[D+c+2] + v.w*u[D+c+3];
      }
    }
    sc[lane] = s;
  }
  __syncthreads();
  vgemm_both<KT>(xt, Pv, bv, vout, lane);
  __syncthreads();
  if (lane < 16) {           // 2-way softmax per dst (q-part cancels)
    const float s0 = sc[2*lane], s1 = sc[2*lane + 1];
    const float mx = fmaxf(s0, s1);
    const float e0 = expf(s0 - mx), e1 = expf(s1 - mx);
    const float inv = 1.0f / (e0 + e1);
    wts[lane] = e0 * inv; wts[16 + lane] = e1 * inv;
  }
  __syncthreads();
#pragma unroll 4
  for (int idx = lane; idx < 16 * D; idx += 32) {
    const int j = idx >> 7, c = idx & (D - 1);
    const float m = wts[j]      * vout[(size_t)(2*j)     * VSTR + c]
                  + wts[16 + j] * vout[(size_t)(2*j + 1) * VSTR + c];
    msgt[(size_t)j * MSTR + c] = f2bf(m);
  }
  __syncthreads();
  const int rem = HALF_LVL - j0;
  gru_write(msgt, MSTR, Pih, bih, gh, ISF ? hf0 : nullptr,
            ISF ? hf : hs, node0, rem < 16 ? rem : 16, lane);
  __syncthreads();
}

__global__ __launch_bounds__(32) void and_level_kernel(
    const int* __restrict__ esrc, const int lvl,
    float* __restrict__ hs, float* __restrict__ hf,
    const float* __restrict__ u_s, const float* __restrict__ u_f,
    const unsigned short* __restrict__ Pv_s, const float* __restrict__ bv_s,
    const unsigned short* __restrict__ Pv_f, const float* __restrict__ bv_f,
    const unsigned short* __restrict__ Pih_s, const float* __restrict__ bih_s,
    const unsigned short* __restrict__ Pih_f, const float* __restrict__ bih_f,
    const float* __restrict__ gh_s, const float* __restrict__ gh_f,
    const float* __restrict__ hf0) {
  __shared__ unsigned short xt[32 * XSTR];
  __shared__ float vout[32 * VSTR];
  __shared__ unsigned short msgt[16 * MSTR];
  __shared__ float sc[32];
  __shared__ float wts[32];
  const int lane = threadIdx.x;
  const int j0 = blockIdx.x * 16;
  int jj = j0 + (lane >> 1); if (jj >= HALF_LVL) jj = HALF_LVL - 1;
  const int src = esrc[(lvl - 1) * EDGES_PER_LVL + 2 * jj + (lane & 1)];
  const int node0 = lvl * N_PER_LEVEL + j0;
  and_phase<false>(lane, j0, node0, src, hs, hf, u_s, Pv_s, bv_s, Pih_s, bih_s, gh_s, hf0,
                   xt, vout, msgt, sc, wts);
  and_phase<true >(lane, j0, node0, src, hs, hf, u_f, Pv_f, bv_f, Pih_f, bih_f, gh_f, hf0,
                   xt, vout, msgt, sc, wts);
}

template<bool ISF>
__device__ __forceinline__ void not_phase(
    int lane, int j0, int node0, int src,
    float* __restrict__ hs, float* __restrict__ hf,
    const unsigned short* __restrict__ Pv, const float* __restrict__ bv,
    const unsigned short* __restrict__ Pih, const float* __restrict__ bih,
    const float* __restrict__ gh, const float* __restrict__ hf0,
    unsigned short* __restrict__ xt, unsigned short* __restrict__ msgt) {
  {  // gather: 2 lanes per row, 64 cols each
    const int row = lane & 15;
    const int c0 = (lane >> 4) * 64;
    const float* p = (ISF ? hf : hs) + (size_t)src * D;
    unsigned short* xr = xt + (size_t)row * NSTR;
#pragma unroll 4
    for (int c = c0; c < c0 + 64; c += 4) {
      float4 v = *(const float4*)(p + c);
      xr[c] = f2bf(v.x); xr[c+1] = f2bf(v.y); xr[c+2] = f2bf(v.z); xr[c+3] = f2bf(v.w);
    }
  }
  __syncthreads();
  {  // msg = X @ Wv^T + bv  (attn == 1 for single-fanin NOT)
    v16bf af[4];
#pragma unroll
    for (int kt = 0; kt < 4; ++kt)
      af[kt] = load_a_frag(xt, NSTR, 0, kt * 32, lane);
    const int g8 = (lane >> 4) << 3;
    const int nlo = lane & 15;
#pragma unroll
    for (int nt = 0; nt < 8; ++nt) {
      v8f acc = {0,0,0,0,0,0,0,0};
#pragma unroll
      for (int kt = 0; kt < 4; ++kt)
        acc = wmma_bf16(af[kt], load_b_frag(Pv, 8, kt, nt, lane), acc);
      const int col = nt * 16 + nlo;
      const float b = bv[col];
#pragma unroll
      for (int v = 0; v < 8; ++v)
        msgt[(size_t)(v + g8) * NSTR + col] = f2bf(acc[v] + b);
    }
  }
  __syncthreads();
  const int rem = HALF_LVL - j0;
  gru_write(msgt, NSTR, Pih, bih, gh, ISF ? hf0 : nullptr,
            ISF ? hf : hs, node0, rem < 16 ? rem : 16, lane);
  __syncthreads();
}

__global__ __launch_bounds__(32) void not_level_kernel(
    const int* __restrict__ esrc, const int lvl,
    float* __restrict__ hs, float* __restrict__ hf,
    const unsigned short* __restrict__ Pv_s, const float* __restrict__ bv_s,
    const unsigned short* __restrict__ Pv_f, const float* __restrict__ bv_f,
    const unsigned short* __restrict__ Pih_s, const float* __restrict__ bih_s,
    const unsigned short* __restrict__ Pih_f, const float* __restrict__ bih_f,
    const float* __restrict__ gh_s, const float* __restrict__ gh_f,
    const float* __restrict__ hf0) {
  __shared__ unsigned short xt[16 * NSTR];
  __shared__ unsigned short msgt[16 * NSTR];
  const int lane = threadIdx.x;
  const int j0 = blockIdx.x * 16;
  int jj = j0 + (lane & 15); if (jj >= HALF_LVL) jj = HALF_LVL - 1;
  const int src = esrc[(lvl - 1) * EDGES_PER_LVL + 2 * HALF_LVL + jj];
  const int node0 = lvl * N_PER_LEVEL + HALF_LVL + j0;
  not_phase<false>(lane, j0, node0, src, hs, hf, Pv_s, bv_s, Pih_s, bih_s, gh_s, hf0, xt, msgt);
  not_phase<true >(lane, j0, node0, src, hs, hf, Pv_f, bv_f, Pih_f, bih_f, gh_f, hf0, xt, msgt);
}

// ---------------- readout ----------------

__global__ __launch_bounds__(256) void readout_kernel(
    const float* __restrict__ hf,
    const float* __restrict__ l1w, const float* __restrict__ l1b,
    const float* __restrict__ g1,  const float* __restrict__ be1,
    const float* __restrict__ l2w, const float* __restrict__ l2b,
    const float* __restrict__ g2,  const float* __restrict__ be2,
    const float* __restrict__ l3w, const float* __restrict__ l3b,
    float* __restrict__ out) {
  __shared__ float W1[32 * D];
  __shared__ float W2[32 * 32];
  __shared__ float B1[32], G1s[32], BE1[32], B2[32], G2s[32], BE2[32], W3[32];
  __shared__ float y1s[256][33];
  const int t = threadIdx.x;
  for (int i = t; i < 32 * D; i += 256) W1[i] = l1w[i];
  for (int i = t; i < 32 * 32; i += 256) W2[i] = l2w[i];
  if (t < 32) { B1[t]=l1b[t]; G1s[t]=g1[t]; BE1[t]=be1[t];
                B2[t]=l2b[t]; G2s[t]=g2[t]; BE2[t]=be2[t]; W3[t]=l3w[t]; }
  __syncthreads();
  const int node = blockIdx.x * 256 + t;
  if (node >= N_NODES) return;
  const float* x = hf + (size_t)node * D;
  float xr[D];
#pragma unroll
  for (int c = 0; c < D; c += 4) {
    float4 v = *(const float4*)(x + c);
    xr[c] = v.x; xr[c+1] = v.y; xr[c+2] = v.z; xr[c+3] = v.w;
  }
  const float inv = 0.99999500003749972f;  // 1/sqrt(1 + 1e-5)
  for (int o = 0; o < 32; ++o) {
    const float* wrow = &W1[o * D];
    float s = 0.0f;
#pragma unroll
    for (int c = 0; c < D; ++c) s += wrow[c] * xr[c];
    y1s[t][o] = fmaxf((s + B1[o]) * G1s[o] * inv + BE1[o], 0.0f);
  }
  float acc = l3b[0];
  for (int o = 0; o < 32; ++o) {
    const float* wrow = &W2[o * 32];
    float s = 0.0f;
#pragma unroll
    for (int c = 0; c < 32; ++c) s += wrow[c] * y1s[t][c];
    acc += W3[o] * fmaxf((s + B2[o]) * G2s[o] * inv + BE2[o], 0.0f);
  }
  out[node] = acc;
}

// ---------------- host ----------------

extern "C" void kernel_launch(void* const* d_in, const int* in_sizes, int n_in,
                              void* d_out, int out_size, void* d_ws, size_t ws_size,
                              hipStream_t stream) {
  (void)in_sizes; (void)n_in; (void)ws_size;
  const int* esrc = (const int*)d_in[0];  // row 0 of edge_index = srcs
#define PF(i) ((const float*)d_in[i])
  // pytree (sorted-key) param leaf indices, after edge_index/gate/forward_level/num_layers_f:
  // and_func: ab4 aw5 kb6 kw7 qb8 qw9 vb10 vw11 | and_strc: ab12 aw13 kb14 kw15 qb16 qw17 vb18 vw19
  // gru_and_func: bhh20 bih21 whh22 wih23 | gru_and_strc: 24..27
  // gru_not_func: 28..31 | gru_not_strc: 32..35 | hf_init: b36 w37
  // mlp: be1 38 be2 39 g1 40 g2 41 l1b 42 l1w 43 l2b 44 l2w 45 l3b 46 l3w 47
  // not_func: ab48 aw49 kb50 kw51 qb52 qw53 vb54 vw55 | not_strc: 56..63

  char* w = (char*)d_ws;
  size_t off = 0;
  auto take = [&](size_t bytes) -> char* {
    char* p = w + off;
    off += (bytes + 255) & ~(size_t)255;
    return p;
  };
  float* hs  = (float*)take((size_t)N_NODES * D * 4);
  float* hf  = (float*)take((size_t)N_NODES * D * 4);
  float* hf0 = (float*)take(D * 4);
  float* u_as = (float*)take(D * 4);
  float* u_af = (float*)take(2 * D * 4);
  float* gh_as = (float*)take(384 * 4);
  float* gh_af = (float*)take(384 * 4);
  float* gh_ns = (float*)take(384 * 4);
  float* gh_nf = (float*)take(384 * 4);
  unsigned short* Pv_as  = (unsigned short*)take((size_t)128 * 128 * 2);
  unsigned short* Pv_af  = (unsigned short*)take((size_t)128 * 256 * 2);
  unsigned short* Pv_ns  = (unsigned short*)take((size_t)128 * 128 * 2);
  unsigned short* Pv_nf  = (unsigned short*)take((size_t)128 * 128 * 2);
  unsigned short* Pih_as = (unsigned short*)take((size_t)384 * 128 * 2);
  unsigned short* Pih_af = (unsigned short*)take((size_t)384 * 128 * 2);
  unsigned short* Pih_ns = (unsigned short*)take((size_t)384 * 128 * 2);
  unsigned short* Pih_nf = (unsigned short*)take((size_t)384 * 128 * 2);

  hf0_kernel<<<1, 128, 0, stream>>>(PF(37), PF(36), hf0);
  init_state_kernel<<<(N_NODES * D) / 256, 256, 0, stream>>>(hs, hf, hf0);
  u_kernel<<<2, 64, 0, stream>>>(PF(15), PF(13), u_as, 128);
  u_kernel<<<4, 64, 0, stream>>>(PF(7),  PF(5),  u_af, 256);
  ghpre_kernel<<<3, 128, 0, stream>>>(PF(26), PF(24), nullptr, gh_as);
  ghpre_kernel<<<3, 128, 0, stream>>>(PF(22), PF(20), hf0,     gh_af);
  ghpre_kernel<<<3, 128, 0, stream>>>(PF(34), PF(32), nullptr, gh_ns);
  ghpre_kernel<<<3, 128, 0, stream>>>(PF(30), PF(28), hf0,     gh_nf);
  pack_b_kernel<<<8 * 4,  32, 0, stream>>>(PF(19), Pv_as, 128, 128);
  pack_b_kernel<<<8 * 8,  32, 0, stream>>>(PF(11), Pv_af, 128, 256);
  pack_b_kernel<<<8 * 4,  32, 0, stream>>>(PF(63), Pv_ns, 128, 128);
  pack_b_kernel<<<8 * 4,  32, 0, stream>>>(PF(55), Pv_nf, 128, 128);
  pack_b_kernel<<<24 * 4, 32, 0, stream>>>(PF(27), Pih_as, 384, 128);
  pack_b_kernel<<<24 * 4, 32, 0, stream>>>(PF(23), Pih_af, 384, 128);
  pack_b_kernel<<<24 * 4, 32, 0, stream>>>(PF(35), Pih_ns, 384, 128);
  pack_b_kernel<<<24 * 4, 32, 0, stream>>>(PF(31), Pih_nf, 384, 128);

  const int blocks = (HALF_LVL + 15) / 16;  // 235
  for (int lvl = 1; lvl < N_LEVELS; ++lvl) {
    and_level_kernel<<<blocks, 32, 0, stream>>>(esrc, lvl, hs, hf, u_as, u_af,
        Pv_as, PF(18), Pv_af, PF(10), Pih_as, PF(25), Pih_af, PF(21), gh_as, gh_af, hf0);
    not_level_kernel<<<blocks, 32, 0, stream>>>(esrc, lvl, hs, hf,
        Pv_ns, PF(62), Pv_nf, PF(54), Pih_ns, PF(33), Pih_nf, PF(29), gh_ns, gh_nf, hf0);
  }
  readout_kernel<<<(N_NODES + 255) / 256, 256, 0, stream>>>(hf,
      PF(43), PF(42), PF(40), PF(38), PF(45), PF(44), PF(41), PF(39),
      PF(47), PF(46), (float*)d_out);
  (void)out_size;
#undef PF
}
